// ExpFilter_48704929137299
// MI455X (gfx1250) — compile-verified, matrix operand order checked by reading, not run on hardware
//
#include <hip/hip_runtime.h>

typedef __attribute__((ext_vector_type(16))) __bf16 v16bf;
typedef __attribute__((ext_vector_type(8)))  float  v8f;

#define T_DIM 2048
#define B_DIM 32
#define D_IN  512
#define D_OUT 512
#define ROWS  (T_DIM * B_DIM)   // 65536

#define BM 128
#define BN 128
#define BK 32
#define KSTEPS (D_IN / BK)      // 16
#define LDS_STRIDE 40           // bf16 elems per row (32 + 8 pad): keeps b128 reads 16B-aligned, spreads banks

#define ALPHA_F 0.36787944117144233f  // exp(-0.001 * 1000)

// Hardware fp32 -> packed bf16 using CDNA5 VOP3P V_FMA_MIXLO/MIXHI_BF16
// (ISA 15.10 opcodes 62/63): d = { bf16(hi), bf16(lo) } in 2 VALU ops.
__device__ __forceinline__ unsigned f2bf_pack(float lo, float hi) {
  unsigned d;
  asm("v_fma_mixlo_bf16 %0, %1, 1.0, 0\n\t"
      "v_fma_mixhi_bf16 %0, %2, 1.0, 0"
      : "=&v"(d)
      : "v"(lo), "v"(hi));
  return d;
}

union FragU {
  v16bf  bf;
  float4 f4[2];
};

struct StageRegs {
  float4 a[4];
  float4 b[4];
};

__device__ __forceinline__ void load_tile(StageRegs& r,
                                          const float* __restrict__ X,
                                          const float* __restrict__ W,
                                          int rowBase, int colBase, int k0, int tid) {
  #pragma unroll
  for (int j = 0; j < 4; ++j) {
    int idx = tid + j * 256;        // 0..1023 float4 slots of a 128x32 tile
    int rr  = idx >> 3;             // 0..127
    int c4  = (idx & 7) << 2;       // 0,4,...,28
    r.a[j] = *(const float4*)(X + (size_t)(rowBase + rr) * D_IN + k0 + c4);
    r.b[j] = *(const float4*)(W + (size_t)(colBase + rr) * D_IN + k0 + c4);
  }
}

__device__ __forceinline__ void store_tile(const StageRegs& r,
                                           unsigned short* sA, unsigned short* sB, int tid) {
  #pragma unroll
  for (int j = 0; j < 4; ++j) {
    int idx = tid + j * 256;
    int rr  = idx >> 3;
    int c4  = (idx & 7) << 2;
    *(uint2*)(&sA[rr * LDS_STRIDE + c4]) =
        make_uint2(f2bf_pack(r.a[j].x, r.a[j].y), f2bf_pack(r.a[j].z, r.a[j].w));
    *(uint2*)(&sB[rr * LDS_STRIDE + c4]) =
        make_uint2(f2bf_pack(r.b[j].x, r.b[j].y), f2bf_pack(r.b[j].z, r.b[j].w));
  }
}

__global__ __launch_bounds__(256)
void gemm_bias_kernel(const float* __restrict__ X,
                      const float* __restrict__ W,
                      const float* __restrict__ bias,
                      float* __restrict__ Y)
{
  // double-buffered bf16 staging: 2 * (5120 + 5120) shorts = 40 KB LDS
  __shared__ __align__(16) unsigned short sA[2][BM * LDS_STRIDE];
  __shared__ __align__(16) unsigned short sB[2][BN * LDS_STRIDE];

  const int tid   = threadIdx.x;
  const int lane  = tid & 31;
  const int wid   = tid >> 5;
  const int waveM = wid & 3;    // 4 waves along M, 32 rows each
  const int waveN = wid >> 2;   // 2 waves along N, 64 cols each

  const int rowBase = blockIdx.y * BM;
  const int colBase = blockIdx.x * BN;

  const int lr   = lane & 15;
  const int koff = (lane < 16) ? 0 : 8;   // 16-bit A/B fragment K-half select

  v8f acc[2][4];
  #pragma unroll
  for (int i = 0; i < 2; ++i)
    #pragma unroll
    for (int j = 0; j < 4; ++j)
      acc[i][j] = 0;

  // ---- pipeline prologue: tile 0 into LDS buf0, tile 1 into registers ----
  StageRegs r;
  load_tile(r, X, W, rowBase, colBase, 0, tid);
  store_tile(r, sA[0], sB[0], tid);
  load_tile(r, X, W, rowBase, colBase, BK, tid);
  __syncthreads();

  #pragma unroll
  for (int kt = 0; kt < KSTEPS; ++kt) {
    const int buf  = kt & 1;
    const int nbuf = buf ^ 1;

    // fragments for this K-step (issue DS loads early)
    FragU fa[2], fb[4];
    #pragma unroll
    for (int tm = 0; tm < 2; ++tm) {
      const unsigned short* p = &sA[buf][(waveM * 32 + tm * 16 + lr) * LDS_STRIDE + koff];
      fa[tm].f4[0] = *(const float4*)(p);
      fa[tm].f4[1] = *(const float4*)(p + 16);
    }
    #pragma unroll
    for (int tn = 0; tn < 4; ++tn) {
      const unsigned short* p = &sB[buf][(waveN * 64 + tn * 16 + lr) * LDS_STRIDE + koff];
      fb[tn].f4[0] = *(const float4*)(p);
      fb[tn].f4[1] = *(const float4*)(p + 16);
    }

    // convert + store the prefetched tile (kt+1) into the other buffer
    if (kt < KSTEPS - 1)
      store_tile(r, sA[nbuf], sB[nbuf], tid);

    // prefetch tile kt+2 from global (overlaps with the WMMAs below)
    if (kt < KSTEPS - 2)
      load_tile(r, X, W, rowBase, colBase, (kt + 2) * BK, tid);

    // 8 WMMAs
    #pragma unroll
    for (int tm = 0; tm < 2; ++tm)
      #pragma unroll
      for (int tn = 0; tn < 4; ++tn)
        acc[tm][tn] = __builtin_amdgcn_wmma_f32_16x16x32_bf16(
            false, fa[tm].bf, false, fb[tn].bf,
            (short)0, acc[tm][tn], false, false);

    __syncthreads();
  }

  // ---- epilogue: bias add + store (C/D layout: VGPR r -> M=r | r+8, lane -> N) ----
  const int mlane = (lane < 16) ? 0 : 8;
  #pragma unroll
  for (int tn = 0; tn < 4; ++tn) {
    int col  = colBase + waveN * 64 + tn * 16 + lr;
    float bv = bias[col];
    #pragma unroll
    for (int tm = 0; tm < 2; ++tm) {
      int row0 = rowBase + waveM * 32 + tm * 16 + mlane;
      #pragma unroll
      for (int rr = 0; rr < 8; ++rr) {
        Y[(size_t)(row0 + rr) * D_OUT + col] = acc[tm][tn][rr] + bv;
      }
    }
  }
}

// In-place temporal IIR: out[t] = alpha*out[t-1] + y[t]; one thread per (b,o) column.
__global__ __launch_bounds__(64)
void expfilter_scan_kernel(float* __restrict__ Y)
{
  const int c = blockIdx.x * 64 + threadIdx.x;   // 0..16383
  const int S = B_DIM * D_OUT;                   // 16384 floats per time step
  float* p = Y + c;
  float acc = 0.0f;                              // t=0: acc = 0*alpha + y[0] = y[0]
  for (int t = 0; t < T_DIM; t += 4) {
    // 4 independent loads issue before the dependent fma chain
    float v0 = p[(size_t)(t + 0) * S];
    float v1 = p[(size_t)(t + 1) * S];
    float v2 = p[(size_t)(t + 2) * S];
    float v3 = p[(size_t)(t + 3) * S];
    acc = fmaf(ALPHA_F, acc, v0); p[(size_t)(t + 0) * S] = acc;
    acc = fmaf(ALPHA_F, acc, v1); p[(size_t)(t + 1) * S] = acc;
    acc = fmaf(ALPHA_F, acc, v2); p[(size_t)(t + 2) * S] = acc;
    acc = fmaf(ALPHA_F, acc, v3); p[(size_t)(t + 3) * S] = acc;
  }
}

extern "C" void kernel_launch(void* const* d_in, const int* in_sizes, int n_in,
                              void* d_out, int out_size, void* d_ws, size_t ws_size,
                              hipStream_t stream) {
  const float* X    = (const float*)d_in[0];   // [T, B, D_IN]
  const float* W    = (const float*)d_in[1];   // [D_OUT, D_IN]
  const float* bias = (const float*)d_in[2];   // [D_OUT]
  float* Y = (float*)d_out;                    // [T, B, D_OUT]

  dim3 grid(D_OUT / BN, ROWS / BM);            // (4, 512)
  gemm_bias_kernel<<<grid, 256, 0, stream>>>(X, W, bias, Y);

  expfilter_scan_kernel<<<(B_DIM * D_OUT) / 64, 64, 0, stream>>>(Y);
}
